// ExpertChoiceMoE_4973572128971
// MI455X (gfx1250) — compile-verified, matrix operand order checked by reading, not run on hardware
//
#include <hip/hip_runtime.h>
#include <hip/hip_bf16.h>
#include <math.h>

// ---------------------------------------------------------------------------
// Expert-choice MoE for MI455X (gfx1250, wave32, WMMA bf16 16x16x32).
// Pipeline: zero -> router(+softmax) -> per-expert bitonic top-k ->
//           gather(bf16) -> GEMM1(xg@w1 +b1, gelu) -> GEMM2(h@w2 +b2,
//           weight * atomic-scatter into results).
// GEMM: block tile 128(m) x 256(n), 8 waves of 64x64 (4x4 WMMA accums),
// double-buffered LDS (one barrier per K-step), B staged as 4x4 micro-tiles
// with K-packed bf16 (v_cvt_pk_bf16_f32 + b64 LDS stores).
// ---------------------------------------------------------------------------

typedef __bf16 bf16_t;
typedef __bf16 v16bf  __attribute__((ext_vector_type(16)));
typedef __bf16 bf16x4 __attribute__((ext_vector_type(4)));
typedef float  v8f    __attribute__((ext_vector_type(8)));
typedef float  f32x4  __attribute__((ext_vector_type(4)));

#define N_TOK 8192   // B*S
#define DM    1024   // model dim D
#define NE    8      // experts
#define FF    4096   // ffn dim F
#define KCAP  1024   // capacity per expert = N/E

__device__ __forceinline__ float gelu_exact(float x) {
  return 0.5f * x * (1.0f + erff(x * 0.70710678118654752f));
}

// ---------------------------------------------------------------------------
// Zero the scatter-add destination (must run every call: graph replays).
// ---------------------------------------------------------------------------
__global__ __launch_bounds__(256) void moe_zero(float* __restrict__ out) {
  int gid = blockIdx.x * 256 + threadIdx.x;
  f32x4 z;
  z[0] = 0.f; z[1] = 0.f; z[2] = 0.f; z[3] = 0.f;
  *(f32x4*)(out + (size_t)gid * 4) = z;
}

// ---------------------------------------------------------------------------
// Router: one wave per token. logits[n][e] = dot(x[n], rw[e]); softmax->probs.
// ---------------------------------------------------------------------------
__global__ __launch_bounds__(256) void moe_router(
    const float* __restrict__ x, const float* __restrict__ rw,
    float* __restrict__ logits, float* __restrict__ probs) {
  const int wave = threadIdx.x >> 5;
  const int lane = threadIdx.x & 31;
  const int tok  = blockIdx.x * 8 + wave;
  const float* xt = x + (size_t)tok * DM;

  float acc[NE];
#pragma unroll
  for (int e = 0; e < NE; ++e) acc[e] = 0.0f;

  for (int it = 0; it < DM / 32; ++it) {
    int d = it * 32 + lane;                 // coalesced across lanes
    float xv = xt[d];
#pragma unroll
    for (int e = 0; e < NE; ++e) acc[e] += xv * rw[e * DM + d];
  }
#pragma unroll
  for (int e = 0; e < NE; ++e)
    for (int m = 16; m > 0; m >>= 1) acc[e] += __shfl_xor(acc[e], m, 32);

  if (lane == 0) {
    float mx = acc[0];
#pragma unroll
    for (int e = 1; e < NE; ++e) mx = fmaxf(mx, acc[e]);
    float ex[NE]; float s = 0.0f;
#pragma unroll
    for (int e = 0; e < NE; ++e) { ex[e] = expf(acc[e] - mx); s += ex[e]; }
    float inv = 1.0f / s;
#pragma unroll
    for (int e = 0; e < NE; ++e) {
      logits[tok * NE + e] = acc[e];
      probs [tok * NE + e] = ex[e] * inv;
    }
  }
}

// ---------------------------------------------------------------------------
// Per-expert top-K: full bitonic sort (desc, idx tiebreak) of 8192 probs in
// LDS (64 KB of 320 KB WGP LDS). One block (1024 thr) per expert.
// ---------------------------------------------------------------------------
__global__ __launch_bounds__(1024) void moe_topk(
    const float* __restrict__ probs, int* __restrict__ sel,
    float* __restrict__ wts, int* __restrict__ selOut) {
  __shared__ float sv[N_TOK];
  __shared__ int   si[N_TOK];
  const int e = blockIdx.x;
  const int tid = threadIdx.x;

  for (int i = tid; i < N_TOK; i += 1024) { sv[i] = probs[i * NE + e]; si[i] = i; }
  __syncthreads();

  for (int k = 2; k <= N_TOK; k <<= 1) {
    for (int j = k >> 1; j > 0; j >>= 1) {
      for (int t = tid; t < N_TOK; t += 1024) {
        int p = t ^ j;
        if (p > t) {
          float v1 = sv[t], v2 = sv[p];
          int   i1 = si[t], i2 = si[p];
          bool desc    = ((t & k) == 0);
          bool inOrder = (v1 > v2) || (v1 == v2 && i1 < i2);
          if (inOrder != desc) { sv[t] = v2; sv[p] = v1; si[t] = i2; si[p] = i1; }
        }
      }
      __syncthreads();
    }
  }
  if (tid < KCAP) {
    sel   [e * KCAP + tid] = si[tid];
    wts   [e * KCAP + tid] = sv[tid];
    selOut[e * KCAP + tid] = si[tid];
  }
}

// ---------------------------------------------------------------------------
// Gather selected tokens to bf16: xg[e][m][d] = bf16(x[sel[e][m]][d]).
// ---------------------------------------------------------------------------
__global__ __launch_bounds__(256) void moe_gather(
    const float* __restrict__ x, const int* __restrict__ sel,
    bf16_t* __restrict__ xg) {
  int idx4 = blockIdx.x * 256 + threadIdx.x;   // 0 .. 2M-1
  int e  = idx4 >> 18;
  int m  = (idx4 >> 8) & (KCAP - 1);
  int d4 = (idx4 & 255) * 4;
  int tok = sel[e * KCAP + m];
  f32x4 v = *(const f32x4*)(x + (size_t)tok * DM + d4);
  bf16x4 o;
#pragma unroll
  for (int q = 0; q < 4; ++q) o[q] = (bf16_t)v[q];
  *(bf16x4*)(xg + (size_t)idx4 * 4) = o;
}

// ---------------------------------------------------------------------------
// Shared GEMM tile machinery.
// Block tile: 128(m) x 256(n), K-step 32, 8 waves (2 x 4) of 64x64 each.
// LDS: double-buffered As[m][k] (128 rows), Bs[n][k] (256 rows), pitch 40
// bf16 = 80 B (16B-aligned rows).
// ---------------------------------------------------------------------------
#define LP   40            // LDS pitch in bf16 elements
#define ASZ  (128 * LP)    // one A buffer
#define BSZ  (256 * LP)    // one B buffer

// Stage A tile (bf16 source, row-major, leading dim LDA): 128x32, b128 copies.
template <int LDA>
__device__ __forceinline__ void stage_a(const bf16_t* __restrict__ Ae, int m0,
                                        int kb, int tid, bf16_t* As) {
#pragma unroll
  for (int p = 0; p < 2; ++p) {
    int c = tid + p * 256;
    int m = c >> 2, ko = (c & 3) * 8;
    *(f32x4*)&As[m * LP + ko] =
        *(const f32x4*)(Ae + (size_t)(m0 + m) * LDA + kb + ko);
  }
}

// Stage B tile (f32 source, row-major [k][n], leading dim LDB) transposed to
// Bs[n][k] bf16, 32(k) x 256(n).  Each thread: two 4x4 (k x n) micro-tiles ->
// b64 LDS stores of K-packed bf16 (enables v_cvt_pk_bf16_f32).
template <int LDB>
__device__ __forceinline__ void stage_b(const float* __restrict__ Be, int n0,
                                        int kb, int tid, bf16_t* Bs) {
#pragma unroll
  for (int p = 0; p < 2; ++p) {
    int c  = tid + p * 256;
    int ng = (c & 63) * 4;   // n offset 0..252
    int kg = (c >> 6) * 4;   // k offset 0..28
    const float* src = Be + (size_t)(kb + kg) * LDB + n0 + ng;
    f32x4 w0 = *(const f32x4*)(src + 0 * LDB);
    f32x4 w1 = *(const f32x4*)(src + 1 * LDB);
    f32x4 w2 = *(const f32x4*)(src + 2 * LDB);
    f32x4 w3 = *(const f32x4*)(src + 3 * LDB);
#pragma unroll
    for (int q = 0; q < 4; ++q) {
      bf16x4 pk;
      pk[0] = (bf16_t)w0[q];
      pk[1] = (bf16_t)w1[q];
      pk[2] = (bf16_t)w2[q];
      pk[3] = (bf16_t)w3[q];
      *(bf16x4*)&Bs[(ng + q) * LP + kg] = pk;
    }
  }
}

union Frag { v16bf v; f32x4 q[2]; };

// Load fragments and run the 4x4 WMMA grid for one K-step (16 WMMAs).
__device__ __forceinline__ void mma_step(const bf16_t* As, const bf16_t* Bs,
                                         int wy, int wx, int lh, int l16,
                                         v8f acc[4][4]) {
  Frag a[4], b[4];
  const int k0 = lh * 8;   // lane-half K split {0..7,16..23} / {8..15,24..31}
#pragma unroll
  for (int i = 0; i < 4; ++i) {
    int row = wy * 64 + i * 16 + l16;
    a[i].q[0] = *(const f32x4*)&As[row * LP + k0];
    a[i].q[1] = *(const f32x4*)&As[row * LP + k0 + 16];
  }
#pragma unroll
  for (int j = 0; j < 4; ++j) {
    int col = wx * 64 + j * 16 + l16;
    b[j].q[0] = *(const f32x4*)&Bs[col * LP + k0];
    b[j].q[1] = *(const f32x4*)&Bs[col * LP + k0 + 16];
  }
#pragma unroll
  for (int i = 0; i < 4; ++i)
#pragma unroll
    for (int j = 0; j < 4; ++j)
      acc[i][j] = __builtin_amdgcn_wmma_f32_16x16x32_bf16(
          false, a[i].v, false, b[j].v, (short)0, acc[i][j], false, false);
}

// ---------------------------------------------------------------------------
// GEMM1: H[e] = gelu(Xg[e](1024x1024 bf16) @ W1[e](1024x4096 f32) + b1[e])
// ---------------------------------------------------------------------------
__global__ __launch_bounds__(256) void moe_gemm1(
    const bf16_t* __restrict__ Xg, const float* __restrict__ W1,
    const float* __restrict__ B1, bf16_t* __restrict__ H) {
  const int e   = blockIdx.z;
  const int n0  = blockIdx.x * 256;
  const int m0  = blockIdx.y * 128;
  const int tid = threadIdx.x;
  const int wave = tid >> 5, lane = tid & 31;
  const int lh = lane >> 4, l16 = lane & 15;
  const int wy = wave >> 2, wx = wave & 3;

  __shared__ bf16_t As[2 * ASZ];
  __shared__ bf16_t Bs[2 * BSZ];

  const bf16_t* Ae = Xg + (size_t)e * KCAP * DM;
  const float*  Be = W1 + (size_t)e * DM * FF;

  v8f acc[4][4];
#pragma unroll
  for (int i = 0; i < 4; ++i)
#pragma unroll
    for (int j = 0; j < 4; ++j)
#pragma unroll
      for (int r = 0; r < 8; ++r) acc[i][j][r] = 0.0f;

  stage_a<DM>(Ae, m0, 0, tid, As);
  stage_b<FF>(Be, n0, 0, tid, Bs);
  __syncthreads();

  int buf = 0;
  for (int kb = 0; kb < DM; kb += 32) {
    int nkb = kb + 32;
    if (nkb < DM) {                       // stage next tile into other buffer
      stage_a<DM>(Ae, m0, nkb, tid, As + (buf ^ 1) * ASZ);
      stage_b<FF>(Be, n0, nkb, tid, Bs + (buf ^ 1) * BSZ);
    }
    mma_step(As + buf * ASZ, Bs + buf * BSZ, wy, wx, lh, l16, acc);
    __syncthreads();
    buf ^= 1;
  }

  bf16_t* He = H + (size_t)e * KCAP * FF;
#pragma unroll
  for (int i = 0; i < 4; ++i)
#pragma unroll
    for (int j = 0; j < 4; ++j) {
      int n = n0 + wx * 64 + j * 16 + l16;
      float bias = B1[e * FF + n];
#pragma unroll
      for (int r = 0; r < 8; ++r) {
        int m = m0 + wy * 64 + i * 16 + r + lh * 8;   // C/D layout: M=r+8*half
        He[(size_t)m * FF + n] = (bf16_t)gelu_exact(acc[i][j][r] + bias);
      }
    }
}

// ---------------------------------------------------------------------------
// GEMM2: Out += weight[e][m] * (H[e](1024x4096 bf16) @ W2[e](4096x1024) + b2)
// scattered with global f32 atomics to results[token].
// ---------------------------------------------------------------------------
__global__ __launch_bounds__(256) void moe_gemm2(
    const bf16_t* __restrict__ H, const float* __restrict__ W2,
    const float* __restrict__ B2, const int* __restrict__ Sel,
    const float* __restrict__ Wts, float* __restrict__ Out) {
  const int e   = blockIdx.z;
  const int n0  = blockIdx.x * 256;
  const int m0  = blockIdx.y * 128;
  const int tid = threadIdx.x;
  const int wave = tid >> 5, lane = tid & 31;
  const int lh = lane >> 4, l16 = lane & 15;
  const int wy = wave >> 2, wx = wave & 3;

  __shared__ bf16_t As[2 * ASZ];
  __shared__ bf16_t Bs[2 * BSZ];

  const bf16_t* Ae = H  + (size_t)e * KCAP * FF;
  const float*  Be = W2 + (size_t)e * FF * DM;

  v8f acc[4][4];
#pragma unroll
  for (int i = 0; i < 4; ++i)
#pragma unroll
    for (int j = 0; j < 4; ++j)
#pragma unroll
      for (int r = 0; r < 8; ++r) acc[i][j][r] = 0.0f;

  stage_a<FF>(Ae, m0, 0, tid, As);
  stage_b<DM>(Be, n0, 0, tid, Bs);
  __syncthreads();

  int buf = 0;
  for (int kb = 0; kb < FF; kb += 32) {
    int nkb = kb + 32;
    if (nkb < FF) {
      stage_a<FF>(Ae, m0, nkb, tid, As + (buf ^ 1) * ASZ);
      stage_b<DM>(Be, n0, nkb, tid, Bs + (buf ^ 1) * BSZ);
    }
    mma_step(As + buf * ASZ, Bs + buf * BSZ, wy, wx, lh, l16, acc);
    __syncthreads();
    buf ^= 1;
  }

  const int*   sel_e = Sel + e * KCAP;
  const float* wts_e = Wts + e * KCAP;
#pragma unroll
  for (int i = 0; i < 4; ++i)
#pragma unroll
    for (int j = 0; j < 4; ++j) {
      int n = n0 + wx * 64 + j * 16 + l16;
      float bias = B2[e * DM + n];
#pragma unroll
      for (int r = 0; r < 8; ++r) {
        int m = m0 + wy * 64 + i * 16 + r + lh * 8;
        int tok = sel_e[m];
        float wg = wts_e[m];
        atomicAdd(&Out[(size_t)tok * DM + n], wg * (acc[i][j][r] + bias));
      }
    }
}

// ---------------------------------------------------------------------------
// Launch
// ---------------------------------------------------------------------------
extern "C" void kernel_launch(void* const* d_in, const int* in_sizes, int n_in,
                              void* d_out, int out_size, void* d_ws, size_t ws_size,
                              hipStream_t stream) {
  (void)in_sizes; (void)n_in; (void)out_size; (void)ws_size;
  const float* x  = (const float*)d_in[0];   // [4,2048,1024]
  const float* rw = (const float*)d_in[1];   // [8,1024]
  const float* w1 = (const float*)d_in[2];   // [8,1024,4096]
  const float* b1 = (const float*)d_in[3];   // [8,4096]
  const float* w2 = (const float*)d_in[4];   // [8,4096,1024]
  const float* b2 = (const float*)d_in[5];   // [8,1024]

  // d_out: results f32 [8192*1024] | logits f32 [8192*8] | sel i32 [8*1024]
  float* outF    = (float*)d_out;
  float* results = outF;
  float* logits  = outF + (size_t)N_TOK * DM;
  int*   selOut  = (int*)(outF + (size_t)N_TOK * DM + (size_t)N_TOK * NE);

  // workspace layout
  char*   ws    = (char*)d_ws;
  float*  probs = (float*)ws;                       // 256 KB
  int*    sel   = (int*)  (ws + 262144);            //  32 KB
  float*  wts   = (float*)(ws + 294912);            //  32 KB
  bf16_t* xg    = (bf16_t*)(ws + 327680);           //  16 MB [8,1024,1024] bf16
  bf16_t* Hbuf  = (bf16_t*)(ws + 17104896);         //  64 MB [8,1024,4096] bf16

  moe_zero  <<<8192, 256, 0, stream>>>(results);
  moe_router<<<1024, 256, 0, stream>>>(x, rw, logits, probs);
  moe_topk  <<<NE, 1024, 0, stream>>>(probs, sel, wts, selOut);
  moe_gather<<<8192, 256, 0, stream>>>(x, sel, xg);
  moe_gemm1 <<<dim3(FF / 256, KCAP / 128, NE), 256, 0, stream>>>(xg, w1, b1, Hbuf);
  moe_gemm2 <<<dim3(DM / 256, KCAP / 128, NE), 256, 0, stream>>>(Hbuf, w2, b2, sel, wts, results);
}
